// MultiTaskDecoder_7765300871256
// MI455X (gfx1250) — matrix operand order repaired and checked by reading, not compiled, not run on hardware
//
#include <hip/hip_runtime.h>
#include <math.h>

// ---------------------------------------------------------------------------
// MI455X (gfx1250) implementation.
// Heavy path = bf16 WMMA (v_wmma_f32_16x16x32_bf16, wave32) for the K/V
// projections and the score GEMM; everything downstream is bandwidth work.
// ---------------------------------------------------------------------------

typedef __attribute__((ext_vector_type(16))) __bf16 v16bf;
typedef __attribute__((ext_vector_type(8)))  __bf16 v8bf;
typedef __attribute__((ext_vector_type(8)))  float  v8f;

constexpr int kB  = 8;
constexpr int kT  = 8;
constexpr int kL  = 4096;
constexpr int kD  = 256;
constexpr int kNH = 4;
constexpr int kHD = 64;
constexpr int kS  = 32768;           // T * L

constexpr int LDSX   = 264;          // padded bf16 row stride (528B = 132 dw -> conflict-free b128)
constexpr int STILE  = 128;          // s-rows per projection block
constexpr int SCHUNK = 2048;         // s per softmax chunk
constexpr int NCHUNK = kS / SCHUNK;  // 16

// workspace layout (bytes)
constexpr size_t OFF_WK  = 0;
constexpr size_t OFF_WV  = OFF_WK  + (size_t)kD * kD * 2;                 // bf16 Wk
constexpr size_t OFF_Q   = OFF_WV  + (size_t)kD * kD * 2;                 // bf16 Wv
constexpr size_t OFF_MLP = OFF_Q   + (size_t)kNH * 16 * kHD * 2;          // bf16 q [h][16][64]
constexpr size_t OFF_ML  = OFF_MLP + (size_t)kB * NCHUNK * 32 * 2 * 4;    // partial (m,l)
constexpr size_t OFF_OP  = OFF_ML  + (size_t)kB * 32 * 2 * 4;             // final (M, 1/L)
constexpr size_t OFF_SC  = OFF_OP  + (size_t)NCHUNK * kB * kT * kD * 4;   // o partials
constexpr size_t OFF_V   = OFF_SC  + (size_t)kB * kS * 32 * 4;            // scores (b,s,h,t) f32

// --- WMMA fragment helpers (contiguous k = 16*(lane/16) + i, same order for A and B, wave32) ----------
// A (16x32 bf16): lane<16 -> row=lane, K {c..c+15}
//                 lane>=16 -> row=lane-16, K {c+16..c+31}
__device__ __forceinline__ v16bf load_a_lds(const __bf16* base, int rowBase,
                                            int colBase, int lane) {
  const int r = rowBase + (lane & 15);
  const int c = colBase + ((lane >> 4) << 4);
  const __bf16* p = base + r * LDSX + c;
  v8bf lo = *(const v8bf*)(p);        // ds_read_b128
  v8bf hi = *(const v8bf*)(p + 8);    // ds_read_b128
  v16bf a;
#pragma unroll
  for (int i = 0; i < 8; ++i) { a[i] = lo[i]; a[i + 8] = hi[i]; }
  return a;
}

// B (32x16 bf16): lane holds column n = lane&15, K = dBase + (lane<16?0:16) .. +15
// (contiguous 32B in a row-major weight matrix W[n][ldb])
__device__ __forceinline__ v16bf load_b_glb(const __bf16* g, int nBase,
                                            int dBase, int lane, int ldb) {
  const int n = nBase + (lane & 15);
  const int d = dBase + ((lane >> 4) << 4);
  return *(const v16bf*)(g + (size_t)n * ldb + d);
}

__device__ __forceinline__ v8f wmma_bf16(v16bf a, v16bf b, v8f c) {
  return __builtin_amdgcn_wmma_f32_16x16x32_bf16(
      /*neg_a=*/false, a, /*neg_b=*/false, b,
      /*c_mod=*/(short)0, c, /*reuse_a=*/false, /*reuse_b=*/false);
}

// ---------------------------------------------------------------------------
// K1: weights -> bf16, q projection (scaled), q pad rows zeroed.
// ---------------------------------------------------------------------------
__global__ void __launch_bounds__(256) k_prep(
    const float* __restrict__ Wk, const float* __restrict__ Wv,
    const float* __restrict__ Wq, const float* __restrict__ bq,
    const float* __restrict__ queries,
    __bf16* __restrict__ wkb, __bf16* __restrict__ wvb, __bf16* __restrict__ qb) {
  const int blk = blockIdx.x, tid = threadIdx.x;
  if (blk < 256) {
    const int i = blk * 256 + tid;
    wkb[i] = (__bf16)Wk[i];
  } else if (blk < 512) {
    const int i = (blk - 256) * 256 + tid;
    wvb[i] = (__bf16)Wv[i];
  } else if (blk < 520) {
    const int t = blk - 512;
    const int n = tid;
    float acc = bq[n];
    for (int d = 0; d < kD; ++d) acc += queries[t * kD + d] * Wq[n * kD + d];
    acc *= 0.125f;  // HD^-0.5
    const int h = n >> 6, dd = n & 63;
    qb[(h * 16 + t) * kHD + dd] = (__bf16)acc;
  } else {  // zero padded t-rows 8..15 of q (B-matrix N padded to 16)
    for (int r = tid; r < kNH * 8 * kHD; r += 256) {
      const int h = r / (8 * kHD);
      const int rem = r % (8 * kHD);
      const int t = 8 + rem / kHD;
      const int d2 = rem % kHD;
      qb[(h * 16 + t) * kHD + d2] = (__bf16)0.f;
    }
  }
}

// ---------------------------------------------------------------------------
// K2: fused K/V projection + raw attention scores. One block = (b, 128 s-rows).
//   LDS: X tile (bf16) + K tile (bf16), both padded stride 264. 135168 B dyn.
// ---------------------------------------------------------------------------
__global__ void __launch_bounds__(256) k_proj_scores(
    const float* __restrict__ enc,
    const __bf16* __restrict__ wkb, const __bf16* __restrict__ wvb,
    const __bf16* __restrict__ qb,
    const float* __restrict__ bk, const float* __restrict__ bv,
    __bf16* __restrict__ vout, float* __restrict__ scores) {
  extern __shared__ __align__(16) char smem[];
  __bf16* X = (__bf16*)smem;            // [128][264]
  __bf16* K = X + STILE * LDSX;         // [128][264]

  const int b = blockIdx.y;
  const int sbase = blockIdx.x * STILE;
  const int tid = threadIdx.x, lane = tid & 31, wave = tid >> 5;

  // stage X tile -> LDS bf16 (coalesced float4 reads)
  const float* src = enc + ((size_t)b * kS + sbase) * kD;
#pragma unroll 4
  for (int it = 0; it < 32; ++it) {
    const int idx = it * 256 + tid;
    const int row = idx >> 6;
    const int col = (idx & 63) << 2;
    float4 f = *(const float4*)(src + row * kD + col);
    __bf16* dst = X + row * LDSX + col;
    dst[0] = (__bf16)f.x; dst[1] = (__bf16)f.y;
    dst[2] = (__bf16)f.z; dst[3] = (__bf16)f.w;
  }
  __syncthreads();

  const int colL = lane & 15;
  const int rbase = wave * 16 + ((lane >> 4) << 3);  // C-layout row base

  // Phase 1: K = X@Wk^T (+bk) -> LDS;  V = X@Wv^T (+bv) -> global ws (bf16)
  for (int nt = 0; nt < 16; ++nt) {
    const int ncol = nt * 16 + colL;
    const float bkf = bk[ncol], bvf = bv[ncol];
    v8f ack = {0.f, 0.f, 0.f, 0.f, 0.f, 0.f, 0.f, 0.f};
    v8f acv = {0.f, 0.f, 0.f, 0.f, 0.f, 0.f, 0.f, 0.f};
#pragma unroll
    for (int dc = 0; dc < 8; ++dc) {
      v16bf a      = load_a_lds(X, wave * 16, dc * 32, lane);
      v16bf bfragk = load_b_glb(wkb, nt * 16, dc * 32, lane, kD);
      v16bf bfragv = load_b_glb(wvb, nt * 16, dc * 32, lane, kD);
      ack = wmma_bf16(a, bfragk, ack);
      acv = wmma_bf16(a, bfragv, acv);
    }
#pragma unroll
    for (int r = 0; r < 8; ++r) {
      const int row = rbase + r;
      K[row * LDSX + ncol] = (__bf16)(ack[r] + bkf);
      vout[((size_t)b * kS + sbase + row) * kD + ncol] = (__bf16)(acv[r] + bvf);
    }
  }
  __syncthreads();

  // Phase 2: scores[b,s,h,t] = sum_d K[s, h*64+d] * q[h][t][d]
  for (int h = 0; h < kNH; ++h) {
    v8f acc = {0.f, 0.f, 0.f, 0.f, 0.f, 0.f, 0.f, 0.f};
#pragma unroll
    for (int dc = 0; dc < 2; ++dc) {
      v16bf a     = load_a_lds(K, wave * 16, h * kHD + dc * 32, lane);
      v16bf bfr_q = load_b_glb(qb, h * 16, dc * 32, lane, kHD);
      acc = wmma_bf16(a, bfr_q, acc);
    }
    if (colL < 8) {  // t < 8 valid; stores only (EXEC was full for the WMMAs)
#pragma unroll
      for (int r = 0; r < 8; ++r) {
        const int s = sbase + rbase + r;
        scores[((size_t)b * kS + s) * 32 + h * 8 + colL] = acc[r];
      }
    }
  }
}

// ---------------------------------------------------------------------------
// K3: per-(b, chunk) masked online max / sum-exp for all 32 (h,t) pairs.
// ---------------------------------------------------------------------------
__global__ void __launch_bounds__(256) k_stats(
    const float* __restrict__ scores, const unsigned char* __restrict__ mask,
    float* __restrict__ mlpart) {
  const int b = blockIdx.y, c = blockIdx.x;
  const int tid = threadIdx.x;
  const int pair = tid & 31, t = pair & 7;
  const int srep = tid >> 5;
  const unsigned char* mb = mask + b * kL;
  float m = -INFINITY, l = 0.f;
  const int s0 = c * SCHUNK;
  for (int k = 0; k < SCHUNK / 8; ++k) {
    const int s = s0 + k * 8 + srep;   // 256 threads cover 8 s x 32 pairs: 1KB coalesced
    if (((s >> 12) <= t) && mb[s & (kL - 1)]) {
      const float sc = scores[((size_t)b * kS + s) * 32 + pair];
      if (sc > m) { l *= __expf(m - sc); m = sc; }
      l += __expf(sc - m);
    }
  }
  __shared__ float sm[256], sl[256];
  sm[tid] = m; sl[tid] = l;
  __syncthreads();
  if (tid < 32) {
    float M = m, L = l;
    for (int j = 1; j < 8; ++j) {
      const float mj = sm[j * 32 + tid], lj = sl[j * 32 + tid];
      const float Mn = fmaxf(M, mj);
      float Ln = 0.f;
      if (L  > 0.f) Ln += L  * __expf(M  - Mn);
      if (lj > 0.f) Ln += lj * __expf(mj - Mn);
      M = Mn; L = Ln;
    }
    mlpart[(((size_t)b * NCHUNK + c) * 32 + tid) * 2 + 0] = M;
    mlpart[(((size_t)b * NCHUNK + c) * 32 + tid) * 2 + 1] = L;
  }
}

// K4: combine chunk partials -> (M, 1/L)
__global__ void __launch_bounds__(256) k_combine(
    const float* __restrict__ mlpart, float* __restrict__ ml) {
  const int tid = threadIdx.x;
  const int b = tid >> 5, pair = tid & 31;
  float M = -INFINITY, L = 0.f;
  for (int c = 0; c < NCHUNK; ++c) {
    const float mj = mlpart[(((size_t)b * NCHUNK + c) * 32 + pair) * 2 + 0];
    const float lj = mlpart[(((size_t)b * NCHUNK + c) * 32 + pair) * 2 + 1];
    const float Mn = fmaxf(M, mj);
    float Ln = 0.f;
    if (L  > 0.f) Ln += L  * __expf(M  - Mn);
    if (lj > 0.f) Ln += lj * __expf(mj - Mn);
    M = Mn; L = Ln;
  }
  ml[(b * 32 + pair) * 2 + 0] = M;
  ml[(b * 32 + pair) * 2 + 1] = (L > 0.f) ? (1.f / L) : 0.f;
}

// ---------------------------------------------------------------------------
// K5: p = exp(s-M)/L on the fly; partial o per chunk (no atomics, deterministic)
//     + attn_w = mean over heads straight to d_out.
// ---------------------------------------------------------------------------
__global__ void __launch_bounds__(256) k_attn(
    const float* __restrict__ scores, const __bf16* __restrict__ v,
    const unsigned char* __restrict__ mask, const float* __restrict__ ml,
    float* __restrict__ opart, float* __restrict__ attnw) {
  const int b = blockIdx.y, c = blockIdx.x;
  const int tid = threadIdx.x;
  __shared__ float sM[32], sR[32];
  if (tid < 32) {
    sM[tid] = ml[(b * 32 + tid) * 2 + 0];
    sR[tid] = ml[(b * 32 + tid) * 2 + 1];
  }
  __syncthreads();

  const unsigned char* mb = mask + b * kL;
  const int pair = tid & 31, h = pair >> 3, t = pair & 7, dg = tid >> 5;
  const float M = sM[pair], R = sR[pair];
  float acc[8] = {0.f, 0.f, 0.f, 0.f, 0.f, 0.f, 0.f, 0.f};
  const int s0 = c * SCHUNK;
  for (int k = 0; k < SCHUNK; ++k) {
    const int s = s0 + k;
    float p = 0.f;
    if (((s >> 12) <= t) && mb[s & (kL - 1)]) {
      const float sc = scores[((size_t)b * kS + s) * 32 + pair];
      p = __expf(sc - M) * R;
    }
    if (p != 0.f) {
      const __bf16* vp = v + ((size_t)b * kS + s) * kD + h * kHD + dg * 8;
      v8bf vv = *(const v8bf*)vp;  // 16B load, L0-shared across t
#pragma unroll
      for (int j = 0; j < 8; ++j) acc[j] += p * (float)vv[j];
    }
  }
  float* op = opart + (((size_t)c * kB + b) * kT + t) * kD + h * kHD + dg * 8;
#pragma unroll
  for (int j = 0; j < 8; ++j) op[j] = acc[j];

  // attn_w[b][t][s] (coalesced 128B stores: 32 consecutive s per t-group)
  const int t2 = tid >> 5, sr = tid & 31;
  for (int k = 0; k < SCHUNK / 32; ++k) {
    const int s = s0 + k * 32 + sr;
    float sum = 0.f;
    if (((s >> 12) <= t2) && mb[s & (kL - 1)]) {
#pragma unroll
      for (int h2 = 0; h2 < kNH; ++h2) {
        const int pr = h2 * 8 + t2;
        sum += __expf(scores[((size_t)b * kS + s) * 32 + pr] - sM[pr]) * sR[pr];
      }
    }
    attnw[((size_t)(b * kT + t2)) * kS + s] = sum * 0.25f;
  }
}

// ---------------------------------------------------------------------------
// K6: reduce o partials, out-proj, LayerNorm, heads, pooled k_log. One block/b.
// ---------------------------------------------------------------------------
__device__ __forceinline__ float block_sum(float vv, float* red, int n) {
  red[n] = vv;
  __syncthreads();
#pragma unroll
  for (int st = 128; st > 0; st >>= 1) {
    if (n < st) red[n] += red[n + st];
    __syncthreads();
  }
  const float r = red[0];
  __syncthreads();
  return r;
}

__global__ void __launch_bounds__(256) k_epilogue(
    const float* __restrict__ opart,
    const float* __restrict__ Wo, const float* __restrict__ bo,
    const float* __restrict__ lng, const float* __restrict__ lnb,
    const float* __restrict__ wloss, const float* __restrict__ bloss,
    const float* __restrict__ winc, const float* __restrict__ binc,
    const float* __restrict__ wkh, const float* __restrict__ bkh,
    float* __restrict__ out) {
  const int b = blockIdx.x, n = threadIdx.x;
  __shared__ float sh[kD];
  __shared__ float red[256];
  float pooled = 0.f;
  for (int t = 0; t < kT; ++t) {
    float o = 0.f;
    for (int c = 0; c < NCHUNK; ++c)
      o += opart[(((size_t)c * kB + b) * kT + t) * kD + n];
    sh[n] = o;
    __syncthreads();
    float acc = bo[n];
    for (int d = 0; d < kD; ++d) acc += Wo[n * kD + d] * sh[d];
    __syncthreads();
    const float mu  = block_sum(acc, red, n) * (1.f / kD);
    const float dd  = acc - mu;
    const float var = block_sum(dd * dd, red, n) * (1.f / kD);
    const float y   = dd * rsqrtf(var + 1e-5f) * lng[n] + lnb[n];
    pooled += y;
    const float lsum = block_sum(y * wloss[n], red, n);
    const float isum = block_sum(y * winc[n],  red, n);
    if (n == 0) {
      out[b * kT + t]      = lsum + bloss[0];   // loss_pred
      out[64 + b * kT + t] = isum + binc[0];    // inc_pred
    }
    __syncthreads();
  }
  const float ks = block_sum((pooled * (1.f / kT)) * wkh[n], red, n);
  if (n == 0) out[128 + b] = ks + bkh[0];       // k_log
}

// ---------------------------------------------------------------------------
extern "C" void kernel_launch(void* const* d_in, const int* in_sizes, int n_in,
                              void* d_out, int out_size, void* d_ws, size_t ws_size,
                              hipStream_t stream) {
  (void)in_sizes; (void)n_in; (void)out_size; (void)ws_size;
  const float* enc          = (const float*)d_in[0];
  const unsigned char* mask = (const unsigned char*)d_in[1];  // jnp bool -> 1 byte
  const float* queries = (const float*)d_in[2];
  const float* Wq  = (const float*)d_in[3];
  const float* bq  = (const float*)d_in[4];
  const float* Wk  = (const float*)d_in[5];
  const float* bk  = (const float*)d_in[6];
  const float* Wv  = (const float*)d_in[7];
  const float* bv  = (const float*)d_in[8];
  const float* Wo  = (const float*)d_in[9];
  const float* bo  = (const float*)d_in[10];
  const float* lng = (const float*)d_in[11];
  const float* lnb = (const float*)d_in[12];
  const float* wloss = (const float*)d_in[13];
  const float* bloss = (const float*)d_in[14];
  const float* winc  = (const float*)d_in[15];
  const float* binc  = (const float*)d_in[16];
  const float* wkh   = (const float*)d_in[17];
  const float* bkh   = (const float*)d_in[18];
  float* out = (float*)d_out;

  char* ws = (char*)d_ws;
  __bf16* wkb   = (__bf16*)(ws + OFF_WK);
  __bf16* wvb   = (__bf16*)(ws + OFF_WV);
  __bf16* qb    = (__bf16*)(ws + OFF_Q);
  float*  mlp   = (float*)(ws + OFF_MLP);
  float*  mlv   = (float*)(ws + OFF_ML);
  float*  opart = (float*)(ws + OFF_OP);
  float*  sc    = (float*)(ws + OFF_SC);
  __bf16* vW    = (__bf16*)(ws + OFF_V);

  k_prep<<<521, 256, 0, stream>>>(Wk, Wv, Wq, bq, queries, wkb, wvb, qb);

  const size_t shbytes = (size_t)2 * STILE * LDSX * sizeof(__bf16);  // 135168 B
  k_proj_scores<<<dim3(kS / STILE, kB), 256, shbytes, stream>>>(
      enc, wkb, wvb, qb, bk, bv, vW, sc);

  k_stats<<<dim3(NCHUNK, kB), 256, 0, stream>>>(sc, mask, mlp);
  k_combine<<<1, 256, 0, stream>>>(mlp, mlv);
  k_attn<<<dim3(NCHUNK, kB), 256, 0, stream>>>(sc, vW, mask, mlv, opart, out + 136);
  k_epilogue<<<kB, 256, 0, stream>>>(opart, Wo, bo, lng, lnb,
                                     wloss, bloss, winc, binc, wkh, bkh, out);
}
